// MSDRescalingLayer_31774168055964
// MI455X (gfx1250) — matrix operand order repaired and checked by reading, compiled, not run on hardware
//
#include <hip/hip_runtime.h>
#include <hip/hip_bf16.h>
#include <math.h>

// ---- problem dims (compile-time) ----
#define BB   16
#define CC   64
#define LL   16384
#define CO   64
#define KK   3
#define PP   1
#define TT   16384              // L + 2P - K + 1
#define LP   (LL + 2 * PP)      // padded length
#define CKD  (CC * KK)          // 192 = GEMM K dim

typedef __attribute__((ext_vector_type(16))) _Float16 v16h;
typedef __attribute__((ext_vector_type(8)))  _Float16 h8;
typedef __attribute__((ext_vector_type(8)))  float    v8f;
typedef __attribute__((ext_vector_type(4)))  float    f4;
typedef __attribute__((ext_vector_type(4)))  int      i4;

#define AST 200   // padded LDS stride (halves); 400 B, 16B-aligned rows
#define BST 200

// CDNA5 async global->LDS path (ASYNCcnt-tracked DMA, bypasses VGPRs).
// Guarded: falls back to a synchronous VGPR copy if the builtin is absent
// (e.g. on the host pass), so compilation never breaks.
#if defined(__gfx1250__) && __has_builtin(__builtin_amdgcn_global_load_async_to_lds_b128)
#define HAVE_ASYNC_LDS 1
// builtin expects int4 pointers: global (AS1) source, LDS (AS3) destination
typedef __attribute__((address_space(1))) i4 gi4;
typedef __attribute__((address_space(3))) i4 li4;
#else
#define HAVE_ASYNC_LDS 0
#endif

// ---------------------------------------------------------------
// Kernel 0: convert conv weight (CO, C, K) f32 -> f16, flat ck = c*K + k
// ---------------------------------------------------------------
__global__ void k_w_to_h(const float* __restrict__ W, _Float16* __restrict__ Wh, int n) {
    int i = blockIdx.x * blockDim.x + threadIdx.x;
    if (i < n) Wh[i] = (_Float16)W[i];
}

// ---------------------------------------------------------------
// Kernel 1: deformable gather+lerp -> LDS A tile (f16), W -> LDS B tile
//           (async DMA overlapped with the gather when available),
//           6x v_wmma_f32_16x16x32_f16 per wave, write y_pre + partial sums.
// grid: (T/16, B), block: 128 (4 waves)
// ---------------------------------------------------------------
__global__ __launch_bounds__(128)
void k_deform_wmma(const float* __restrict__ x,
                   const float* __restrict__ off,
                   const _Float16* __restrict__ Wh,
                   const float* __restrict__ bias,
                   float* __restrict__ y_pre,
                   float* __restrict__ partials) {
    __shared__ __align__(16) _Float16 A_lds[16 * AST];
    __shared__ __align__(16) _Float16 B_lds[64 * BST];
    __shared__ int   p0s[48];
    __shared__ int   p1s[48];
    __shared__ float wls[48];
    __shared__ float red[256];

    const int t0  = blockIdx.x * 16;
    const int b   = blockIdx.y;
    const int tid = threadIdx.x;

    // --- stage W (f16) into LDS, row o (padded stride), contiguous ck ---
#if HAVE_ASYNC_LDS
    // 64 rows x 24 chunks of 16 B each = 1536 async B128 transfers; DMA runs
    // in the background while we do the deformable gather below.
    for (int e = tid; e < 64 * 24; e += 128) {
        int row = e / 24, chk = e % 24;
        void* g = (void*)(Wh + row * CKD + chk * 8);
        void* l = (void*)((char*)&B_lds[0] + row * (BST * 2) + chk * 16);
        __builtin_amdgcn_global_load_async_to_lds_b128(
            (gi4*)g, (li4*)l, /*offset=*/0, /*cpol=*/0);
    }
#else
    for (int e = tid; e < CO * CKD; e += 128)
        B_lds[(e / CKD) * BST + (e % CKD)] = Wh[e];
#endif

    // --- sampling positions: 16 t-rows x 3 taps (offsets indep. of channel) ---
    if (tid < 48) {
        int m = tid / KK, k = tid % KK;
        int t = t0 + m;
        float pos = (float)t + (float)k + off[(b * TT + t) * KK + k];
        pos = fminf(fmaxf(pos, 0.0f), (float)(LP - 1));
        int p0 = (int)floorf(pos);
        int p1 = min(p0 + 1, LP - 1);
        p0s[tid] = p0;
        p1s[tid] = p1;
        wls[tid] = pos - (float)p0;
    }
    __syncthreads();   // p0s/p1s/wls visible (W DMA still in flight if async)

    // --- gather + linear interp -> A tile (f16); overlaps with W DMA ---
    const float* xb = x + (size_t)b * CC * LL;
    for (int e = tid; e < 16 * CKD; e += 128) {
        int m  = e / CKD;
        int ck = e % CKD;
        int c  = ck / KK, k = ck % KK;
        int s  = m * KK + k;
        int p0 = p0s[s], p1 = p1s[s];
        float w  = wls[s];
        float v0 = (p0 >= 1 && p0 <= LL) ? xb[c * LL + (p0 - 1)] : 0.0f;  // zero pad
        float v1 = (p1 >= 1 && p1 <= LL) ? xb[c * LL + (p1 - 1)] : 0.0f;
        A_lds[m * AST + ck] = (_Float16)((1.0f - w) * v0 + w * v1);
    }

#if HAVE_ASYNC_LDS
#if __has_builtin(__builtin_amdgcn_s_wait_asynccnt)
    __builtin_amdgcn_s_wait_asynccnt(0);
#else
    asm volatile("s_wait_asynccnt 0x0" ::: "memory");
#endif
#endif
    __syncthreads();   // A tile + W tile both resident

    // --- WMMA: D[m][n] = sum_ck A[m][ck] * W[n][ck] ---
    const int lane = tid & 31;
    const int wv   = tid >> 5;
    const int n0   = wv * 16;         // this wave's N tile
    const int col  = lane & 15;
    const int hi   = lane >> 4;       // lane half selects K sub-block

    v8f acc = {};
#pragma unroll
    for (int kb = 0; kb < CKD; kb += 32) {
        // A 16x32 f16 layout: lanes 0-15 K={0..7,16..23}, lanes 16-31 K={8..15,24..31}
        h8 alo = *(const h8*)&A_lds[col * AST + kb + hi * 8];
        h8 ahi = *(const h8*)&A_lds[col * AST + kb + 16 + hi * 8];
        // B 32x16 f16 layout: lanes 0-15 K={0..7,16..23}, lanes 16-31 K={8..15,24..31}; col = N
        h8 blo = *(const h8*)&B_lds[(n0 + col) * BST + kb + hi * 8];
        h8 bhi = *(const h8*)&B_lds[(n0 + col) * BST + kb + 16 + hi * 8];
        v16h a  = __builtin_shufflevector(alo, ahi, 0,1,2,3,4,5,6,7,8,9,10,11,12,13,14,15);
        v16h bf = __builtin_shufflevector(blo, bhi, 0,1,2,3,4,5,6,7,8,9,10,11,12,13,14,15);
        acc = __builtin_amdgcn_wmma_f32_16x16x32_f16(
                  /*neg_a=*/false, a, /*neg_b=*/false, bf,
                  /*c_mod=*/(short)0, acc, /*reuse_a=*/false, /*reuse_b=*/false);
    }

    // --- epilogue: bias, store (8 contiguous t per lane), partial sums ---
    const int   o    = n0 + col;
    const float bo   = bias[o];
    float* dst = y_pre + ((size_t)b * CO + o) * TT + (t0 + hi * 8);
    float lsum = 0.0f, lsq = 0.0f;
    f4 v0, v1;
#pragma unroll
    for (int r = 0; r < 4; ++r) { float v = acc[r]     + bo; v0[r] = v; lsum += v; lsq += v * v; }
#pragma unroll
    for (int r = 0; r < 4; ++r) { float v = acc[4 + r] + bo; v1[r] = v; lsum += v; lsq += v * v; }
    *(f4*)(dst)     = v0;
    *(f4*)(dst + 4) = v1;

    red[tid] = lsum; red[128 + tid] = lsq;
    __syncthreads();
    for (int s = 64; s > 0; s >>= 1) {
        if (tid < s) { red[tid] += red[tid + s]; red[128 + tid] += red[128 + tid + s]; }
        __syncthreads();
    }
    if (tid == 0) {
        int bid = blockIdx.y * gridDim.x + blockIdx.x;
        partials[2 * bid]     = red[0];
        partials[2 * bid + 1] = red[128];
    }
}

// ---------------------------------------------------------------
// Kernel 2: deterministic reduction of per-block partials -> mu, 1/sigma
// ---------------------------------------------------------------
__global__ void k_finalize(const float* __restrict__ partials, int nblk,
                           float* __restrict__ stats) {
    __shared__ float s[512];
    int tid = threadIdx.x;
    float a = 0.0f, c = 0.0f;
    for (int i = tid; i < nblk; i += 256) { a += partials[2 * i]; c += partials[2 * i + 1]; }
    s[tid] = a; s[256 + tid] = c;
    __syncthreads();
    for (int st = 128; st > 0; st >>= 1) {
        if (tid < st) { s[tid] += s[tid + st]; s[256 + tid] += s[256 + tid + st]; }
        __syncthreads();
    }
    if (tid == 0) {
        const float n  = (float)BB * (float)CO * (float)TT;
        float mu  = s[0] / n;
        float var = s[256] / n - mu * mu;
        stats[0] = mu;
        stats[1] = rsqrtf(var + 1e-5f);
    }
}

// ---------------------------------------------------------------
// Kernel 3: LayerNorm affine + channel concat + maxpool3(pad=-inf) + v+sin^2
// grid: (T/TS, CO+C, B), block 256. LDS strip with 1-element halo.
// ---------------------------------------------------------------
#define TS 2048
__global__ __launch_bounds__(256)
void k_ln_pool_act(const float* __restrict__ x,
                   const float* __restrict__ y_pre,
                   const float* __restrict__ ln_w,
                   const float* __restrict__ ln_b,
                   const float* __restrict__ stats,
                   float* __restrict__ out) {
    __shared__ float z[TS + 2];
    const int tbase = blockIdx.x * TS;
    const int ch    = blockIdx.y;
    const int b     = blockIdx.z;
    const int tid   = threadIdx.x;
    const float mu = stats[0], rs = stats[1];

    if (ch < CO) {
        const float* yr = y_pre + ((size_t)b * CO + ch) * TT;
        const float* lw = ln_w  + ((size_t)b * CO + ch) * TT;
        const float* lb = ln_b  + ((size_t)b * CO + ch) * TT;
        __builtin_prefetch(yr + tbase, 0, 0);   // global_prefetch_b8
        for (int j = tid; j < TS + 2; j += 256) {
            int t = tbase + j - 1;
            z[j] = (t < 0 || t >= TT) ? -INFINITY
                                      : (yr[t] - mu) * rs * lw[t] + lb[t];
        }
    } else {
        const float* xr = x + ((size_t)b * CC + (ch - CO)) * LL;
        for (int j = tid; j < TS + 2; j += 256) {
            int t = tbase + j - 1;
            z[j] = (t < 0 || t >= TT) ? -INFINITY : xr[t];
        }
    }
    __syncthreads();

    float* orow = out + ((size_t)b * (CO + CC) + ch) * TT + tbase;
    for (int j = tid; j < TS; j += 256) {
        float v = fmaxf(z[j], fmaxf(z[j + 1], z[j + 2]));
        float sn = sinf(v);
        orow[j] = v + sn * sn;
    }
}

// ---------------------------------------------------------------
extern "C" void kernel_launch(void* const* d_in, const int* in_sizes, int n_in,
                              void* d_out, int out_size, void* d_ws, size_t ws_size,
                              hipStream_t stream) {
    (void)in_sizes; (void)n_in; (void)out_size; (void)ws_size;
    const float* x    = (const float*)d_in[0];   // (B, C, L)
    const float* off  = (const float*)d_in[1];   // (B, 1, T, K)
    const float* W    = (const float*)d_in[2];   // (CO, C, K)
    const float* bias = (const float*)d_in[3];   // (CO,)
    const float* ln_w = (const float*)d_in[4];   // (B, CO, T)
    const float* ln_b = (const float*)d_in[5];   // (B, CO, T)
    float* out = (float*)d_out;                  // (B, CO+C, T)

    // workspace layout
    char* ws = (char*)d_ws;
    float*     stats    = (float*)ws;                       // 2 floats
    _Float16*  Wh       = (_Float16*)(ws + 64);             // 24576 B
    float*     partials = (float*)(ws + 32768);             // 16384 blocks * 2 f32 = 128 KB
    float*     y_pre    = (float*)(ws + (size_t)(256 * 1024)); // B*CO*T f32 = 64 MB

    const int nW = CO * CC * KK;                            // 12288
    k_w_to_h<<<(nW + 255) / 256, 256, 0, stream>>>(W, Wh, nW);

    dim3 g1(TT / 16, BB);                                   // (1024, 16)
    k_deform_wmma<<<g1, 128, 0, stream>>>(x, off, Wh, bias, y_pre, partials);

    k_finalize<<<1, 256, 0, stream>>>(partials, (TT / 16) * BB, stats);

    dim3 g3(TT / TS, CO + CC, BB);                          // (8, 128, 16)
    k_ln_pool_act<<<g3, 256, 0, stream>>>(x, y_pre, ln_w, ln_b, stats, out);
}
